// CustomComplexSSM_798863917308
// MI455X (gfx1250) — matrix operand order repaired and checked by reading, not compiled, not run on hardware
//
#include <hip/hip_runtime.h>
#include <hip/hip_bf16.h>

// ---------------------------------------------------------------------------
// CustomComplexSSM on gfx1250 (MI455X):
//   RMSNorm -> (un @ B) via bf16 WMMA -> chunked linear scan (diagonal A)
//   -> Re(x @ C) + D via bf16 WMMA.  FFT conv replaced by O(L) scan.
// ---------------------------------------------------------------------------

typedef __attribute__((ext_vector_type(8)))  float  v8f;
typedef __attribute__((ext_vector_type(16))) __bf16 v16bf;
typedef __attribute__((ext_vector_type(8)))  __bf16 v8bf;

#define DM      512            // D_IN == D_ST == D_OUT
#define SEQ     4096
#define BATCH   4
#define MROWS   (BATCH * SEQ)  // 16384
#define CHUNK   128
#define NCHUNK  (SEQ / CHUNK)  // 32

// ---- workspace layout (bytes) ----------------------------------------------
static const size_t SZ_UN  = (size_t)MROWS * DM * 2;           // bf16  16 MiB
static const size_t SZ_MAT = (size_t)DM * DM * 2;              // bf16 512 KiB
static const size_t SZ_F   = (size_t)MROWS * DM * 4;           // f32   32 MiB
static const size_t SZ_ST  = (size_t)BATCH * NCHUNK * DM * 2 * 4; // 512 KiB

static const size_t OFF_UN  = 0;
static const size_t OFF_BTR = OFF_UN  + SZ_UN;
static const size_t OFF_BTI = OFF_BTR + SZ_MAT;
static const size_t OFF_CTR = OFF_BTI + SZ_MAT;
static const size_t OFF_CTN = OFF_CTR + SZ_MAT;   // holds -Im(C)
static const size_t OFF_UBR = OFF_CTN + SZ_MAT;   // f32, becomes x_re in-place
static const size_t OFF_UBI = OFF_UBR + SZ_F;     // f32, becomes x_im in-place
static const size_t OFF_XBR = OFF_UBI + SZ_F;     // bf16 copy of x_re
static const size_t OFF_XBI = OFF_XBR + SZ_UN;    // bf16 copy of x_im
static const size_t OFF_ST  = OFF_XBI + SZ_UN;    // chunk-final states
static const size_t OFF_CA  = OFF_ST  + SZ_ST;    // chunk-incoming carries

// ---------------------------------------------------------------------------
// 1) RMSNorm + cast to bf16.  One 256-thread block per row of 512.
// ---------------------------------------------------------------------------
__global__ void k_rmsnorm(const float* __restrict__ u, const float* __restrict__ w,
                          __bf16* __restrict__ un) {
  __shared__ float red[8];
  const int row = blockIdx.x;
  const int tid = threadIdx.x;                // 0..255
  const float* up = u + (size_t)row * DM;
  float a0 = up[tid], a1 = up[tid + 256];
  float s = a0 * a0 + a1 * a1;
#pragma unroll
  for (int off = 16; off > 0; off >>= 1) s += __shfl_down(s, off, 32);
  if ((tid & 31) == 0) red[tid >> 5] = s;
  __syncthreads();
  if (tid < 32) {
    float v = (tid < 8) ? red[tid] : 0.0f;
#pragma unroll
    for (int off = 4; off > 0; off >>= 1) v += __shfl_down(v, off, 32);
    if (tid == 0) red[0] = v;
  }
  __syncthreads();
  const float rs = rsqrtf(red[0] * (1.0f / DM) + 1e-6f);
  __bf16* op = un + (size_t)row * DM;
  op[tid]       = (__bf16)(a0 * rs * w[tid]);
  op[tid + 256] = (__bf16)(a1 * rs * w[tid + 256]);
}

// ---------------------------------------------------------------------------
// 2) Split complex64 B/C into transposed bf16 matrices [N][K] (B-operand form).
//    Ct_neg_im holds -Im(C) so the output GEMM is pure accumulation.
// ---------------------------------------------------------------------------
__global__ void k_prep(const float* __restrict__ Bc, const float* __restrict__ Cc,
                       __bf16* __restrict__ btr, __bf16* __restrict__ bti,
                       __bf16* __restrict__ ctr, __bf16* __restrict__ ctn) {
  const int idx = blockIdx.x * blockDim.x + threadIdx.x;  // 0..262143
  const int k = idx >> 9, n = idx & 511;
  const size_t src = (size_t)2 * ((size_t)k * DM + n);
  const size_t dst = (size_t)n * DM + k;
  if (blockIdx.y == 0) {
    btr[dst] = (__bf16)Bc[src];
    bti[dst] = (__bf16)Bc[src + 1];
  } else {
    ctr[dst] = (__bf16)Cc[src];
    ctn[dst] = (__bf16)(-Cc[src + 1]);
  }
}

// ---------------------------------------------------------------------------
// WMMA fragment loaders.
// A (16x32 bf16, row-major [M][K]): lanes 0-15 hold K {0..15}
// of row M=lane; lanes 16-31 hold K {16..31} (matches B).  Two b128 loads.
// ---------------------------------------------------------------------------
__device__ __forceinline__ v16bf load_a_frag(const __bf16* __restrict__ p, int koff) {
  v8bf lo = *(const v8bf*)(p + koff);        // K: koff .. koff+7
  v8bf hi = *(const v8bf*)(p + koff + 8);    // K: koff+8 .. koff+15
  v16bf r;
#pragma unroll
  for (int i = 0; i < 8; ++i) { r[i] = lo[i]; r[i + 8] = hi[i]; }
  return r;
}
// B (32x16 bf16) from transposed [N][K]: lane holds column N = lane&15,
// K range = half*16 + 0..15 contiguous -> single 32B load.
__device__ __forceinline__ v16bf load_b_frag(const __bf16* __restrict__ bt,
                                             int col, int k0, int half) {
  return *(const v16bf*)(bt + (size_t)col * DM + k0 + half * 16);
}

// ---------------------------------------------------------------------------
// 3) uB = un @ B (re and im selected by blockIdx.z).  Wave tile = 16x64.
//    grid (128, 8, 2), block 256 (8 waves).  K=512 in steps of 32.
// ---------------------------------------------------------------------------
__global__ void k_gemm_uB(const __bf16* __restrict__ un,
                          const __bf16* __restrict__ btr,
                          const __bf16* __restrict__ bti,
                          float* __restrict__ ubr, float* __restrict__ ubi) {
  const __bf16* bt  = blockIdx.z ? bti : btr;
  float*        out = blockIdx.z ? ubi : ubr;
  const int wave = threadIdx.x >> 5, lane = threadIdx.x & 31;
  const int half = lane >> 4, l16 = lane & 15;
  const int row0 = (blockIdx.x * 8 + wave) * 16;
  const int col0 = blockIdx.y * 64;
  const __bf16* ap = un + (size_t)(row0 + l16) * DM;
  v8f acc0 = {}, acc1 = {}, acc2 = {}, acc3 = {};
  for (int k0 = 0; k0 < DM; k0 += 32) {
    v16bf a  = load_a_frag(ap, k0 + half * 16);
    v16bf b0 = load_b_frag(bt, col0 +  0 + l16, k0, half);
    v16bf b1 = load_b_frag(bt, col0 + 16 + l16, k0, half);
    v16bf b2 = load_b_frag(bt, col0 + 32 + l16, k0, half);
    v16bf b3 = load_b_frag(bt, col0 + 48 + l16, k0, half);
    acc0 = __builtin_amdgcn_wmma_f32_16x16x32_bf16(false, a, false, b0, (short)0, acc0, false, false);
    acc1 = __builtin_amdgcn_wmma_f32_16x16x32_bf16(false, a, false, b1, (short)0, acc1, false, false);
    acc2 = __builtin_amdgcn_wmma_f32_16x16x32_bf16(false, a, false, b2, (short)0, acc2, false, false);
    acc3 = __builtin_amdgcn_wmma_f32_16x16x32_bf16(false, a, false, b3, (short)0, acc3, false, false);
  }
  const int orow = row0 + 8 * half;   // C/D layout: VGPR r -> M = r + 8*half, N = l16
#pragma unroll
  for (int r = 0; r < 8; ++r) {
    float* op = out + (size_t)(orow + r) * DM + col0 + l16;
    op[0] = acc0[r]; op[16] = acc1[r]; op[32] = acc2[r]; op[48] = acc3[r];
  }
}

// ---------------------------------------------------------------------------
// 4) Scan phase 1: per (b, chunk, z) local scan with zero init, in place.
//    65536 lanes; z fastest -> coalesced.  Store chunk-final state.
// ---------------------------------------------------------------------------
__global__ void k_scan1(float* __restrict__ xr, float* __restrict__ xi,
                        const float* __restrict__ Ac, float* __restrict__ st) {
  const int gid = blockIdx.x * blockDim.x + threadIdx.x;
  const int z = gid & 511, chunk = (gid >> 9) & (NCHUNK - 1), b = gid >> 14;
  const float ar = Ac[2 * z], ai = Ac[2 * z + 1];
  const size_t base = ((size_t)(b * SEQ + chunk * CHUNK)) * DM + z;
  float sr = 0.0f, si = 0.0f;
  for (int t = 0; t < CHUNK; ++t) {
    const size_t p = base + (size_t)t * DM;
    const float cr = xr[p], ci = xi[p];
    const float nr = ar * sr - ai * si + cr;
    const float ni = ar * si + ai * sr + ci;
    xr[p] = nr; xi[p] = ni; sr = nr; si = ni;
  }
  const size_t so = (size_t)2 * (((size_t)b * NCHUNK + chunk) * DM + z);
  st[so] = sr; st[so + 1] = si;
}

// ---------------------------------------------------------------------------
// 5) Scan phase 2: serial combine over 32 chunks per (b,z); A^128 by squaring.
//    ca[c] = state entering chunk c.
// ---------------------------------------------------------------------------
__global__ void k_scan2(const float* __restrict__ Ac, const float* __restrict__ st,
                        float* __restrict__ ca) {
  const int gid = blockIdx.x * blockDim.x + threadIdx.x;  // 2048
  const int z = gid & 511, b = gid >> 9;
  float pr = Ac[2 * z], pi = Ac[2 * z + 1];
#pragma unroll
  for (int i = 0; i < 7; ++i) {                  // A^(2^7) = A^CHUNK
    const float nr = pr * pr - pi * pi;
    const float ni = 2.0f * pr * pi;
    pr = nr; pi = ni;
  }
  float cr = 0.0f, ci = 0.0f;
  for (int c = 0; c < NCHUNK; ++c) {
    const size_t o = (size_t)2 * (((size_t)b * NCHUNK + c) * DM + z);
    ca[o] = cr; ca[o + 1] = ci;
    const float sr = st[o], si = st[o + 1];
    const float nr = pr * cr - pi * ci + sr;
    const float ni = pr * ci + pi * cr + si;
    cr = nr; ci = ni;
  }
}

// ---------------------------------------------------------------------------
// 6) Scan phase 3: x[t] = local[t] + A^(t-t0+1) * carry.  Writes f32 back,
//    bf16 copy for the WMMA output GEMM, and the final state x[:,L-1].
// ---------------------------------------------------------------------------
__global__ void k_scan3(float* __restrict__ xr, float* __restrict__ xi,
                        const float* __restrict__ Ac, const float* __restrict__ ca,
                        __bf16* __restrict__ xbr, __bf16* __restrict__ xbi,
                        float* __restrict__ ylast) {
  const int gid = blockIdx.x * blockDim.x + threadIdx.x;
  const int z = gid & 511, chunk = (gid >> 9) & (NCHUNK - 1), b = gid >> 14;
  const float ar = Ac[2 * z], ai = Ac[2 * z + 1];
  const size_t co = (size_t)2 * (((size_t)b * NCHUNK + chunk) * DM + z);
  const float Xr = ca[co], Xi = ca[co + 1];
  const size_t base = ((size_t)(b * SEQ + chunk * CHUNK)) * DM + z;
  float apr = ar, api = ai;     // A^1, A^2, ...
  float fr = 0.0f, fi = 0.0f;
  for (int t = 0; t < CHUNK; ++t) {
    const size_t p = base + (size_t)t * DM;
    const float vr = xr[p] + (apr * Xr - api * Xi);
    const float vi = xi[p] + (apr * Xi + api * Xr);
    xr[p] = vr; xi[p] = vi;
    xbr[p] = (__bf16)vr; xbi[p] = (__bf16)vi;
    const float nr = apr * ar - api * ai;
    const float ni = apr * ai + api * ar;
    apr = nr; api = ni;
    fr = vr; fi = vi;
  }
  if (chunk == NCHUNK - 1) {    // x[:, -1] as interleaved complex64
    ylast[2 * (b * DM + z)]     = fr;
    ylast[2 * (b * DM + z) + 1] = fi;
  }
}

// ---------------------------------------------------------------------------
// 7) y = x_re@Re(C) + x_im@(-Im(C)) + D.  Same tiling, two WMMA per K-step.
//    grid (128, 8), block 256.
// ---------------------------------------------------------------------------
__global__ void k_gemm_y(const __bf16* __restrict__ xbr, const __bf16* __restrict__ xbi,
                         const __bf16* __restrict__ ctr, const __bf16* __restrict__ ctn,
                         const float* __restrict__ Dv, float* __restrict__ y) {
  const int wave = threadIdx.x >> 5, lane = threadIdx.x & 31;
  const int half = lane >> 4, l16 = lane & 15;
  const int row0 = (blockIdx.x * 8 + wave) * 16;
  const int col0 = blockIdx.y * 64;
  const __bf16* apr = xbr + (size_t)(row0 + l16) * DM;
  const __bf16* api = xbi + (size_t)(row0 + l16) * DM;
  v8f acc0 = {}, acc1 = {}, acc2 = {}, acc3 = {};
  for (int k0 = 0; k0 < DM; k0 += 32) {
    v16bf arf = load_a_frag(apr, k0 + half * 16);
    v16bf aif = load_a_frag(api, k0 + half * 16);
    v16bf br0 = load_b_frag(ctr, col0 +  0 + l16, k0, half);
    v16bf br1 = load_b_frag(ctr, col0 + 16 + l16, k0, half);
    v16bf br2 = load_b_frag(ctr, col0 + 32 + l16, k0, half);
    v16bf br3 = load_b_frag(ctr, col0 + 48 + l16, k0, half);
    acc0 = __builtin_amdgcn_wmma_f32_16x16x32_bf16(false, arf, false, br0, (short)0, acc0, false, false);
    acc1 = __builtin_amdgcn_wmma_f32_16x16x32_bf16(false, arf, false, br1, (short)0, acc1, false, false);
    acc2 = __builtin_amdgcn_wmma_f32_16x16x32_bf16(false, arf, false, br2, (short)0, acc2, false, false);
    acc3 = __builtin_amdgcn_wmma_f32_16x16x32_bf16(false, arf, false, br3, (short)0, acc3, false, false);
    v16bf bi0 = load_b_frag(ctn, col0 +  0 + l16, k0, half);
    v16bf bi1 = load_b_frag(ctn, col0 + 16 + l16, k0, half);
    v16bf bi2 = load_b_frag(ctn, col0 + 32 + l16, k0, half);
    v16bf bi3 = load_b_frag(ctn, col0 + 48 + l16, k0, half);
    acc0 = __builtin_amdgcn_wmma_f32_16x16x32_bf16(false, aif, false, bi0, (short)0, acc0, false, false);
    acc1 = __builtin_amdgcn_wmma_f32_16x16x32_bf16(false, aif, false, bi1, (short)0, acc1, false, false);
    acc2 = __builtin_amdgcn_wmma_f32_16x16x32_bf16(false, aif, false, bi2, (short)0, acc2, false, false);
    acc3 = __builtin_amdgcn_wmma_f32_16x16x32_bf16(false, aif, false, bi3, (short)0, acc3, false, false);
  }
  const int orow = row0 + 8 * half;
  const float d0 = Dv[col0 + l16], d1 = Dv[col0 + 16 + l16];
  const float d2 = Dv[col0 + 32 + l16], d3 = Dv[col0 + 48 + l16];
#pragma unroll
  for (int r = 0; r < 8; ++r) {
    float* op = y + (size_t)(orow + r) * DM + col0 + l16;
    op[0] = acc0[r] + d0; op[16] = acc1[r] + d1;
    op[32] = acc2[r] + d2; op[48] = acc3[r] + d3;
  }
}

// ---------------------------------------------------------------------------
extern "C" void kernel_launch(void* const* d_in, const int* in_sizes, int n_in,
                              void* d_out, int out_size, void* d_ws, size_t ws_size,
                              hipStream_t stream) {
  (void)in_sizes; (void)n_in; (void)out_size; (void)ws_size;
  const float* u  = (const float*)d_in[0];   // [4,4096,512] f32
  const float* Ac = (const float*)d_in[1];   // [512] complex64 -> float pairs
  const float* Bc = (const float*)d_in[2];   // [512,512] complex64
  const float* Cc = (const float*)d_in[3];   // [512,512] complex64
  const float* Dv = (const float*)d_in[4];   // [512] f32
  const float* wn = (const float*)d_in[5];   // [512] f32

  char* ws = (char*)d_ws;
  __bf16* un  = (__bf16*)(ws + OFF_UN);
  __bf16* btr = (__bf16*)(ws + OFF_BTR);
  __bf16* bti = (__bf16*)(ws + OFF_BTI);
  __bf16* ctr = (__bf16*)(ws + OFF_CTR);
  __bf16* ctn = (__bf16*)(ws + OFF_CTN);
  float*  ubr = (float*)(ws + OFF_UBR);
  float*  ubi = (float*)(ws + OFF_UBI);
  __bf16* xbr = (__bf16*)(ws + OFF_XBR);
  __bf16* xbi = (__bf16*)(ws + OFF_XBI);
  float*  st  = (float*)(ws + OFF_ST);
  float*  ca  = (float*)(ws + OFF_CA);

  float* y     = (float*)d_out;                       // [4,4096,512] f32
  float* ylast = (float*)d_out + (size_t)MROWS * DM;  // [4,512] complex64 tail

  k_rmsnorm<<<MROWS, 256, 0, stream>>>(u, wn, un);
  k_prep<<<dim3(1024, 2), 256, 0, stream>>>(Bc, Cc, btr, bti, ctr, ctn);
  k_gemm_uB<<<dim3(128, 8, 2), 256, 0, stream>>>(un, btr, bti, ubr, ubi);
  k_scan1<<<(BATCH * NCHUNK * DM) / 256, 256, 0, stream>>>(ubr, ubi, Ac, st);
  k_scan2<<<(BATCH * DM) / 256, 256, 0, stream>>>(Ac, st, ca);
  k_scan3<<<(BATCH * NCHUNK * DM) / 256, 256, 0, stream>>>(ubr, ubi, Ac, ca, xbr, xbi, ylast);
  k_gemm_y<<<dim3(128, 8), 256, 0, stream>>>(xbr, xbi, ctr, ctn, Dv, y);
}
